// DGD_69140383531126
// MI455X (gfx1250) — matrix operand order repaired and checked by reading, not compiled, not run on hardware
//
#include <hip/hip_runtime.h>
#include <hip/hip_bf16.h>
#include <cmath>
#include <cstdint>

// Problem constants (match reference setup_inputs)
#define B_SZ   4096
#define D_SZ   512
#define H_SZ   2048
#define K_MIX  10
#define R_SMP  2
#define T_STEPS 3
#define N_TOT  (R_SMP * B_SZ)          // 8192
#define LOG2PI_F 1.8378770664093453f
#define LR_F   0.05f
#define INV_TEMP 10.0f                 // 1/0.1

typedef __attribute__((ext_vector_type(16))) __bf16 v16bf;
typedef __attribute__((ext_vector_type(8)))  float  v8f;

// ---------- small helpers ----------
static __device__ __forceinline__ unsigned short f2bf(float f) {
  union { float f; unsigned u; } v; v.f = f;
  unsigned r = v.u + 0x7FFFu + ((v.u >> 16) & 1u);
  return (unsigned short)(r >> 16);
}
static __device__ __forceinline__ float bf2f(unsigned short h) {
  union { unsigned u; float f; } v; v.u = ((unsigned)h) << 16;
  return v.f;
}
static __device__ __forceinline__ unsigned pcg_hash(unsigned v) {
  v = v * 747796405u + 2891336453u;
  unsigned w = ((v >> ((v >> 28u) + 4u)) ^ v) * 277803737u;
  return (w >> 22u) ^ w;
}
static __device__ __forceinline__ float u01(unsigned h) {
  return ((float)(h >> 8) + 0.5f) * (1.0f / 16777216.0f); // (0,1)
}
// block = 256 threads
static __device__ __forceinline__ float blockReduceMax(float v, float* red) {
  int t = threadIdx.x;
  red[t] = v; __syncthreads();
  for (int s = 128; s > 0; s >>= 1) { if (t < s) red[t] = fmaxf(red[t], red[t + s]); __syncthreads(); }
  float r = red[0]; __syncthreads();
  return r;
}
static __device__ __forceinline__ float blockReduceSum(float v, float* red) {
  int t = threadIdx.x;
  red[t] = v; __syncthreads();
  for (int s = 128; s > 0; s >>= 1) { if (t < s) red[t] += red[t + s]; __syncthreads(); }
  float r = red[0]; __syncthreads();
  return r;
}

// ---------- CDNA5 async-to-LDS primitives ----------
// LDS destination = LDS base + per-lane VGPR offset; low 32 bits of a generic __shared__ address
// ARE the wave-relative LDS offset (flat aperture).
static __device__ __forceinline__ void async_b128(unsigned lds_off, const void* gaddr) {
  asm volatile("global_load_async_to_lds_b128 %0, %1, off"
               :: "v"(lds_off), "v"((unsigned long long)(uintptr_t)gaddr)
               : "memory");
}
static __device__ __forceinline__ void wait_async_le12() {
  asm volatile("s_wait_asynccnt 12" ::: "memory");
}
static __device__ __forceinline__ void wait_async_le0() {
  asm volatile("s_wait_asynccnt 0" ::: "memory");
}

// ---------- WMMA GEMM: C(MxN) = A(MxK) * Bt(NxK)^T (+ epilogue) ----------
// Block tile 128x64, 4 waves x (32x64 wave tile: 2 M-frags x 4 N-frags).
// LDS double-buffered, K-tile = 64 (2 WMMA k-steps/stage), async-to-LDS fill.
// EPI 0: c = relu(acc + bias[n])           -> Cb (bf16)   [fwd h]
// EPI 1: c = acc + bias[n]                 -> Cf (f32)    [fwd xr]
// EPI 2: c = acc * (mask_ref[row,col] > 0) -> Cb (bf16)   [bwd dH]
// EPI 3: c = acc                           -> Cf (f32)    [bwd dz]
#define KTILE      64
#define ROWPITCH   144u                       // 128B data + 16B pad (bank skew)
#define A_BUFSZ    (128u * ROWPITCH)          // 18432 B
#define B_BUFSZ    (64u * ROWPITCH)           //  9216 B
#define SMEM_BYTES (2u * A_BUFSZ + 2u * B_BUFSZ) // 55296 B
#define GEMM_TPB   128                        // 4 waves

template<int EPI>
__global__ __launch_bounds__(GEMM_TPB)
void gemm_wmma_bf16(const unsigned short* __restrict__ A,
                    const unsigned short* __restrict__ Bt,
                    const float* __restrict__ bias,
                    const unsigned short* __restrict__ mask_ref,
                    float* __restrict__ Cf,
                    unsigned short* __restrict__ Cb,
                    int M, int N, int Kd)
{
  __shared__ unsigned char smem[SMEM_BYTES];

  const unsigned tid = threadIdx.x;
  const int lane    = tid & 31;
  const int wave    = tid >> 5;          // 0..3
  const int halfsel = (lane >> 4) & 1;   // 0: lanes 0-15, 1: lanes 16-31
  const int l15     = lane & 15;
  const int mBlock  = blockIdx.y * 128;
  const int mBase   = mBlock + wave * 32;   // wave owns 32 rows
  const int nBase   = blockIdx.x * 64;

  const unsigned smem_lds = (unsigned)(uintptr_t)(void*)smem;
  const unsigned aOfs[2] = { 0u, A_BUFSZ };
  const unsigned bOfs[2] = { 2u * A_BUFSZ, 2u * A_BUFSZ + B_BUFSZ };

  v8f zero = {};
  v8f acc[2][4];
#pragma unroll
  for (int mi = 0; mi < 2; ++mi)
#pragma unroll
    for (int j = 0; j < 4; ++j) acc[mi][j] = zero;

  union FragU { v16bf v; uint4 q[2]; };

  // stage filler: A tile 128x64 bf16 (1024 x 16B), B tile 64x64 bf16 (512 x 16B)
  // 128 threads -> 8 A chunks + 4 B chunks = 12 async b128 per thread per stage
  auto issue = [&](int buf, int kk) {
#pragma unroll
    for (int i = 0; i < 8; ++i) {
      unsigned c = tid + (unsigned)GEMM_TPB * i;   // 0..1023
      unsigned row = c >> 3, j = c & 7u;
      const unsigned short* g = A + (size_t)(mBlock + (int)row) * Kd + (kk + (int)j * 8);
      async_b128(smem_lds + aOfs[buf] + row * ROWPITCH + j * 16u, g);
    }
#pragma unroll
    for (int i = 0; i < 4; ++i) {
      unsigned c = tid + (unsigned)GEMM_TPB * i;   // 0..511
      unsigned row = c >> 3, j = c & 7u;
      const unsigned short* g = Bt + (size_t)(nBase + (int)row) * Kd + (kk + (int)j * 8);
      async_b128(smem_lds + bOfs[buf] + row * ROWPITCH + j * 16u, g);
    }
  };

  const int ntiles = Kd / KTILE;
  issue(0, 0);
  for (int kt = 0; kt < ntiles; ++kt) {
    const int buf = kt & 1;
    if (kt + 1 < ntiles) {
      issue(buf ^ 1, (kt + 1) * KTILE);  // stream next stage into other buffer
      wait_async_le12();                 // retire this stage's 12 loads (in-order)
    } else {
      wait_async_le0();
    }
    __syncthreads();                     // stage visible to all waves

    // Fragment base addresses in LDS for this wave/lane.
    const unsigned char* sA  = smem + aOfs[buf] + (unsigned)(wave * 32 + l15) * ROWPITCH
                                    + (unsigned)(halfsel * 32);
    const unsigned char* sB0 = smem + bOfs[buf] + (unsigned)l15 * ROWPITCH
                                    + (unsigned)(halfsel * 32);
#pragma unroll
    for (int ks = 0; ks < 2; ++ks) {     // two k-steps of 32 per stage
      // Load ALL fragments of this k-step into fresh registers first, so the
      // scheduler can batch the ds_loads under a single s_wait_dscnt.
      FragU a0, a1, b0, b1, b2, b3;
      a0.q[0] = *(const uint4*)(sA + ks * 64);
      a0.q[1] = *(const uint4*)(sA + ks * 64 + 16);
      a1.q[0] = *(const uint4*)(sA + 16u * ROWPITCH + ks * 64);
      a1.q[1] = *(const uint4*)(sA + 16u * ROWPITCH + ks * 64 + 16);
      b0.q[0] = *(const uint4*)(sB0 + 0u * ROWPITCH + ks * 64);
      b0.q[1] = *(const uint4*)(sB0 + 0u * ROWPITCH + ks * 64 + 16);
      b1.q[0] = *(const uint4*)(sB0 + 16u * ROWPITCH + ks * 64);
      b1.q[1] = *(const uint4*)(sB0 + 16u * ROWPITCH + ks * 64 + 16);
      b2.q[0] = *(const uint4*)(sB0 + 32u * ROWPITCH + ks * 64);
      b2.q[1] = *(const uint4*)(sB0 + 32u * ROWPITCH + ks * 64 + 16);
      b3.q[0] = *(const uint4*)(sB0 + 48u * ROWPITCH + ks * 64);
      b3.q[1] = *(const uint4*)(sB0 + 48u * ROWPITCH + ks * 64 + 16);

      acc[0][0] = __builtin_amdgcn_wmma_f32_16x16x32_bf16(false, a0.v, false, b0.v, (short)0, acc[0][0], false, false);
      acc[0][1] = __builtin_amdgcn_wmma_f32_16x16x32_bf16(false, a0.v, false, b1.v, (short)0, acc[0][1], false, false);
      acc[0][2] = __builtin_amdgcn_wmma_f32_16x16x32_bf16(false, a0.v, false, b2.v, (short)0, acc[0][2], false, false);
      acc[0][3] = __builtin_amdgcn_wmma_f32_16x16x32_bf16(false, a0.v, false, b3.v, (short)0, acc[0][3], false, false);
      acc[1][0] = __builtin_amdgcn_wmma_f32_16x16x32_bf16(false, a1.v, false, b0.v, (short)0, acc[1][0], false, false);
      acc[1][1] = __builtin_amdgcn_wmma_f32_16x16x32_bf16(false, a1.v, false, b1.v, (short)0, acc[1][1], false, false);
      acc[1][2] = __builtin_amdgcn_wmma_f32_16x16x32_bf16(false, a1.v, false, b2.v, (short)0, acc[1][2], false, false);
      acc[1][3] = __builtin_amdgcn_wmma_f32_16x16x32_bf16(false, a1.v, false, b3.v, (short)0, acc[1][3], false, false);
    }
    __syncthreads();                     // reads done before next stage overwrites
  }

  // Epilogue. C layout: VGPR r -> row mBase + mi*16 + r + 8*halfsel, col nBase + j*16 + l15
#pragma unroll
  for (int mi = 0; mi < 2; ++mi) {
#pragma unroll
    for (int j = 0; j < 4; ++j) {
      const int col = nBase + j * 16 + l15;
      float bv = 0.0f;
      if (EPI == 0 || EPI == 1) bv = bias[col];
#pragma unroll
      for (int r = 0; r < 8; ++r) {
        const int row = mBase + mi * 16 + r + halfsel * 8;
        const size_t idx = (size_t)row * N + col;
        float v = acc[mi][j][r] + bv;
        if (EPI == 0) {
          v = v > 0.0f ? v : 0.0f;
          Cb[idx] = f2bf(v);
        } else if (EPI == 1) {
          Cf[idx] = v;
        } else if (EPI == 2) {
          float href = bf2f(mask_ref[idx]);
          Cb[idx] = f2bf(href > 0.0f ? v : 0.0f);
        } else {
          Cf[idx] = v;
        }
      }
    }
  }
}

// ---------- prep kernels ----------
__global__ void k_weights_prep(const float* __restrict__ W1, const float* __restrict__ W2,
                               unsigned short* __restrict__ W1bf, unsigned short* __restrict__ W1t,
                               unsigned short* __restrict__ W2bf, unsigned short* __restrict__ W2t)
{
  size_t i = (size_t)blockIdx.x * blockDim.x + threadIdx.x;
  if (i >= (size_t)D_SZ * H_SZ) return;
  // W1 is (D,H) row-major
  int d1 = (int)(i / H_SZ), h1 = (int)(i % H_SZ);
  unsigned short w1 = f2bf(W1[i]);
  W1bf[i] = w1;                               // (D,H): Bt for dz = dH @ W1^T
  W1t[(size_t)h1 * D_SZ + d1] = w1;           // (H,D): Bt for h  = z  @ W1
  // W2 is (H,D) row-major
  int h2 = (int)(i / D_SZ), d2 = (int)(i % D_SZ);
  unsigned short w2 = f2bf(W2[i]);
  W2bf[i] = w2;                               // (H,D): Bt for dH = gxr @ W2^T
  W2t[(size_t)d2 * H_SZ + h2] = w2;           // (D,H): Bt for xr = h   @ W2
}

__global__ void k_prior_prep(const float* __restrict__ plogvar, const float* __restrict__ pmix,
                             float* __restrict__ stdv, float* __restrict__ stdinv,
                             float* __restrict__ invvar, float* __restrict__ cterm)
{
  int i = blockIdx.x * blockDim.x + threadIdx.x;
  if (i >= K_MIX * D_SZ) return;
  float lv = plogvar[i];
  int k = i / D_SZ;
  stdv[i]   = expf(0.5f * lv);
  stdinv[i] = expf(-0.5f * lv);
  invvar[i] = expf(-lv);
  // -log(std) - 0.5*log(2pi) + log(mix_k)
  cterm[i] = -0.5f * lv - 0.5f * LOG2PI_F + logf(pmix[k]);
}

__global__ void k_zero(float* __restrict__ m, float* __restrict__ v, size_t n) {
  size_t i = (size_t)blockIdx.x * blockDim.x + threadIdx.x;
  if (i < n) { m[i] = 0.0f; v[i] = 0.0f; }
}

__global__ void k_mixw(const float* __restrict__ pmix, float* __restrict__ wmix) {
  int n = blockIdx.x * blockDim.x + threadIdx.x;
  if (n >= N_TOT) return;
  float lg[K_MIX]; float mx = -1e30f;
#pragma unroll
  for (int k = 0; k < K_MIX; ++k) {
    unsigned h = pcg_hash(0x6A09E667u ^ (unsigned)(n * K_MIX + k));
    float u = u01(h);
    float g = -logf(-logf(u));                // gumbel
    float l = (pmix[k] + g) * INV_TEMP;       // reference uses raw mix, not log mix
    lg[k] = l; mx = fmaxf(mx, l);
  }
  float s = 0.0f;
#pragma unroll
  for (int k = 0; k < K_MIX; ++k) { lg[k] = expf(lg[k] - mx); s += lg[k]; }
  float inv = 1.0f / s;
#pragma unroll
  for (int k = 0; k < K_MIX; ++k) wmix[(size_t)n * K_MIX + k] = lg[k] * inv;
}

__global__ void k_init_z(const float* __restrict__ wmix, const float* __restrict__ stdv,
                         const float* __restrict__ pmean, float* __restrict__ z)
{
  size_t i = (size_t)blockIdx.x * blockDim.x + threadIdx.x;
  if (i >= (size_t)N_TOT * D_SZ) return;
  int n = (int)(i / D_SZ), d = (int)(i % D_SZ);
  const float* wn = wmix + (size_t)n * K_MIX;
  float acc = 0.0f;
#pragma unroll
  for (int k = 0; k < K_MIX; ++k) {
    unsigned idx = (unsigned)(((size_t)n * K_MIX + k) * D_SZ + d);
    unsigned ha = pcg_hash(0xBB67AE85u ^ idx);
    unsigned hb = pcg_hash(ha + 0x3C6EF372u);
    float u1 = u01(ha), u2 = u01(hb);
    float e = sqrtf(-2.0f * logf(u1)) * cosf(6.28318530718f * u2); // N(0,1)
    acc += wn[k] * (e * stdv[k * D_SZ + d] + pmean[k * D_SZ + d]);
  }
  z[i] = acc;
}

// ---------- per-step elementwise ----------
__global__ void k_f2bf(const float* __restrict__ in, unsigned short* __restrict__ outb, size_t n) {
  size_t i = (size_t)blockIdx.x * blockDim.x + threadIdx.x;
  if (i < n) outb[i] = f2bf(in[i]);
}

// g_xr = d(sum ll)/dxr - (1/N)*(z - xr)   (stored bf16 for backward GEMM)
__global__ void k_xr_grad(const float* __restrict__ x, const float* __restrict__ xr,
                          const float* __restrict__ z, unsigned short* __restrict__ dXrbf,
                          float invN)
{
  size_t i = (size_t)blockIdx.x * blockDim.x + threadIdx.x;
  if (i >= (size_t)N_TOT * D_SZ) return;
  int n = (int)(i / D_SZ), d = (int)(i % D_SZ);
  float xv  = x[(size_t)(n % B_SZ) * D_SZ + d];
  float xrv = xr[i];
  float zv  = z[i];
  float g = 0.0f;
  if (xrv > 1e-12f && xrv < 1.0f - 1e-12f) {      // clip gradient gate
    g = xv / xrv - (1.0f - xv) / (1.0f - xrv);    // d ll / d xr
  }
  g -= invN * (zv - xrv);                         // -(1/N) * d lq / d xr
  dXrbf[i] = f2bf(g);
}

// prior: lpz_n = sum_k LSE_d(a_nkd); gz += (1/N) * d lpz / dz. One block per n.
__global__ __launch_bounds__(256)
void k_prior(const float* __restrict__ z, const float* __restrict__ pmean,
             const float* __restrict__ stdinv, const float* __restrict__ invvar,
             const float* __restrict__ cterm, float* __restrict__ gz,
             float* __restrict__ lpz, float invN)
{
  __shared__ float red[256];
  const int n = blockIdx.x;
  const int t = threadIdx.x;
  const int d0 = t, d1 = t + 256;
  const size_t base = (size_t)n * D_SZ;
  const float z0 = z[base + d0], z1 = z[base + d1];
  float acc0 = 0.0f, acc1 = 0.0f, lpzsum = 0.0f;
  for (int k = 0; k < K_MIX; ++k) {
    const int kb = k * D_SZ;
    float mu0 = pmean[kb + d0], mu1 = pmean[kb + d1];
    float s0 = stdinv[kb + d0], s1 = stdinv[kb + d1];
    float t0 = (z0 - mu0) * s0, t1 = (z1 - mu1) * s1;
    float a0 = -0.5f * t0 * t0 + cterm[kb + d0];
    float a1 = -0.5f * t1 * t1 + cterm[kb + d1];
    float amax = blockReduceMax(fmaxf(a0, a1), red);
    float e0 = expf(a0 - amax), e1 = expf(a1 - amax);
    float sum = blockReduceSum(e0 + e1, red);
    float inv = 1.0f / sum;
    acc0 += (e0 * inv) * (-(z0 - mu0) * invvar[kb + d0]);
    acc1 += (e1 * inv) * (-(z1 - mu1) * invvar[kb + d1]);
    lpzsum += amax + logf(sum);
  }
  gz[base + d0] += invN * acc0;
  gz[base + d1] += invN * acc1;
  if (t == 0) lpz[n] = lpzsum;
}

// last-step per-sample ELBO (pre-update z): elbo = ll + lpz - lq
__global__ __launch_bounds__(256)
void k_persample(const float* __restrict__ x, const float* __restrict__ xr,
                 const float* __restrict__ z, const float* __restrict__ lpz,
                 float* __restrict__ elbo)
{
  __shared__ float red[256];
  const int n = blockIdx.x;
  const int t = threadIdx.x;
  const float* xrow = x + (size_t)(n % B_SZ) * D_SZ;
  const size_t base = (size_t)n * D_SZ;
  float llsum = 0.0f, sqsum = 0.0f;
  for (int d = t; d < D_SZ; d += 256) {
    float xv = xrow[d], xrv = xr[base + d], zv = z[base + d];
    float p = fminf(fmaxf(xrv, 1e-12f), 1.0f - 1e-12f);
    float lp  = fmaxf(logf(p), -100.0f);
    float l1p = fmaxf(log1pf(-p), -100.0f);
    llsum += xv * lp + (1.0f - xv) * l1p;
    float dz = zv - xrv; sqsum += dz * dz;
  }
  float ll = blockReduceSum(llsum, red);
  float sq = blockReduceSum(sqsum, red);
  if (t == 0) elbo[n] = ll + lpz[n] + 0.5f * sq + 0.5f * LOG2PI_F * (float)D_SZ;
}

__global__ void k_adam(float* __restrict__ z, const float* __restrict__ gz,
                       const float* __restrict__ xr, float* __restrict__ m,
                       float* __restrict__ v, float invN, float bc1inv, float bc2inv)
{
  size_t i = (size_t)blockIdx.x * blockDim.x + threadIdx.x;
  if (i >= (size_t)N_TOT * D_SZ) return;
  float zv = z[i];
  float g = gz[i] + invN * (zv - xr[i]);        // + (1/N) * direct dlq/dz term
  float mv = 0.9f * m[i] + 0.1f * g;
  float vv = 0.999f * v[i] + 0.001f * g * g;
  m[i] = mv; v[i] = vv;
  float mhat = mv * bc1inv, vhat = vv * bc2inv;
  z[i] = zv - LR_F * mhat / (sqrtf(vhat) + 1e-8f);
}

// literal z_rb[:, idx, :] gather (idx in {0,1} used as batch index, as written)
__global__ void k_select(const float* __restrict__ xr, const float* __restrict__ z,
                         const float* __restrict__ elbo, float* __restrict__ out)
{
  const size_t RBD = (size_t)R_SMP * B_SZ * D_SZ;
  size_t i = (size_t)blockIdx.x * blockDim.x + threadIdx.x;
  if (i >= RBD) return;
  int d = (int)(i % D_SZ);
  size_t rb = i / D_SZ;
  int b = (int)(rb % B_SZ);
  int r = (int)(rb / B_SZ);
  float e0 = elbo[b], e1 = elbo[B_SZ + b];
  int idxv = (e1 > e0) ? 1 : 0;                   // argmax (first on tie)
  size_t src = ((size_t)r * B_SZ + (size_t)idxv) * D_SZ + d;
  out[i]       = xr[src];                         // xr_sel
  out[RBD + i] = z[src];                          // z_sel
  if (r == 0 && d == 0) out[2 * RBD + b] = fmaxf(e0, e1); // elbo_max
}

// ---------- launcher ----------
extern "C" void kernel_launch(void* const* d_in, const int* in_sizes, int n_in,
                              void* d_out, int out_size, void* d_ws, size_t ws_size,
                              hipStream_t stream)
{
  (void)in_sizes; (void)n_in; (void)out_size; (void)ws_size;
  const float* x       = (const float*)d_in[0];
  const float* W1      = (const float*)d_in[1];
  const float* b1      = (const float*)d_in[2];
  const float* W2      = (const float*)d_in[3];
  const float* b2      = (const float*)d_in[4];
  const float* pmean   = (const float*)d_in[5];
  const float* plogvar = (const float*)d_in[6];
  const float* pmix    = (const float*)d_in[7];
  // d_in[8] = R (=2), d_in[9] = T (=3): device scalars, compiled in.

  const size_t ND = (size_t)N_TOT * D_SZ;
  const size_t NH = (size_t)N_TOT * H_SZ;
  const size_t DH = (size_t)D_SZ * H_SZ;
  const size_t KD = (size_t)K_MIX * D_SZ;
  const float invN = 1.0f / (float)N_TOT;

  // workspace arena (256B aligned)
  char* p = (char*)d_ws;
  auto alloc = [&](size_t bytes) -> void* {
    void* r = (void*)p;
    p += (bytes + 255) & ~(size_t)255;
    return r;
  };
  float* z      = (float*)alloc(ND * 4);
  float* m      = (float*)alloc(ND * 4);
  float* v      = (float*)alloc(ND * 4);
  float* xr     = (float*)alloc(ND * 4);
  float* gz     = (float*)alloc(ND * 4);
  unsigned short* zbf    = (unsigned short*)alloc(ND * 2);
  unsigned short* dXrbf  = (unsigned short*)alloc(ND * 2);
  unsigned short* hbf    = (unsigned short*)alloc(NH * 2);
  unsigned short* dHbf   = (unsigned short*)alloc(NH * 2);
  unsigned short* W1bf   = (unsigned short*)alloc(DH * 2);
  unsigned short* W1t    = (unsigned short*)alloc(DH * 2);
  unsigned short* W2bf   = (unsigned short*)alloc(DH * 2);
  unsigned short* W2t    = (unsigned short*)alloc(DH * 2);
  float* stdv   = (float*)alloc(KD * 4);
  float* stdinv = (float*)alloc(KD * 4);
  float* invvar = (float*)alloc(KD * 4);
  float* cterm  = (float*)alloc(KD * 4);
  float* wmix   = (float*)alloc((size_t)N_TOT * K_MIX * 4);
  float* lpz    = (float*)alloc((size_t)N_TOT * 4);
  float* elbo   = (float*)alloc((size_t)N_TOT * 4);

  const int TPB = 256;
  // prep
  k_weights_prep<<<(unsigned)((DH + TPB - 1) / TPB), TPB, 0, stream>>>(W1, W2, W1bf, W1t, W2bf, W2t);
  k_prior_prep<<<(unsigned)((KD + TPB - 1) / TPB), TPB, 0, stream>>>(plogvar, pmix, stdv, stdinv, invvar, cterm);
  k_zero<<<(unsigned)((ND + TPB - 1) / TPB), TPB, 0, stream>>>(m, v, ND);
  k_mixw<<<(N_TOT + TPB - 1) / TPB, TPB, 0, stream>>>(pmix, wmix);
  k_init_z<<<(unsigned)((ND + TPB - 1) / TPB), TPB, 0, stream>>>(wmix, stdv, pmean, z);

  for (int t = 1; t <= T_STEPS; ++t) {
    k_f2bf<<<(unsigned)((ND + TPB - 1) / TPB), TPB, 0, stream>>>(z, zbf, ND);
    // h = relu(z @ W1 + b1)
    gemm_wmma_bf16<0><<<dim3(H_SZ / 64, N_TOT / 128), GEMM_TPB, 0, stream>>>(
        zbf, W1t, b1, nullptr, nullptr, hbf, N_TOT, H_SZ, D_SZ);
    // xr = h @ W2 + b2
    gemm_wmma_bf16<1><<<dim3(D_SZ / 64, N_TOT / 128), GEMM_TPB, 0, stream>>>(
        hbf, W2t, b2, nullptr, xr, nullptr, N_TOT, D_SZ, H_SZ);
    // g_xr
    k_xr_grad<<<(unsigned)((ND + TPB - 1) / TPB), TPB, 0, stream>>>(x, xr, z, dXrbf, invN);
    // dH = relu'(h) * (g_xr @ W2^T)
    gemm_wmma_bf16<2><<<dim3(H_SZ / 64, N_TOT / 128), GEMM_TPB, 0, stream>>>(
        dXrbf, W2bf, nullptr, hbf, nullptr, dHbf, N_TOT, H_SZ, D_SZ);
    // gz = dH @ W1^T
    gemm_wmma_bf16<3><<<dim3(D_SZ / 64, N_TOT / 128), GEMM_TPB, 0, stream>>>(
        dHbf, W1bf, nullptr, nullptr, gz, nullptr, N_TOT, D_SZ, H_SZ);
    // + (1/N)*dlpz/dz, lpz per sample
    k_prior<<<N_TOT, TPB, 0, stream>>>(z, pmean, stdinv, invvar, cterm, gz, lpz, invN);
    if (t == T_STEPS) {
      k_persample<<<N_TOT, TPB, 0, stream>>>(x, xr, z, lpz, elbo);
    }
    float bc1inv = (float)(1.0 / (1.0 - pow(0.9, (double)t)));
    float bc2inv = (float)(1.0 / (1.0 - pow(0.999, (double)t)));
    k_adam<<<(unsigned)((ND + TPB - 1) / TPB), TPB, 0, stream>>>(z, gz, xr, m, v, invN, bc1inv, bc2inv);
  }

  const size_t RBD = (size_t)R_SMP * B_SZ * D_SZ;
  k_select<<<(unsigned)((RBD + TPB - 1) / TPB), TPB, 0, stream>>>(xr, z, elbo, (float*)d_out);
}
